// SutraV051_87892210745387
// MI455X (gfx1250) — matrix operand order repaired and checked by reading, not compiled, not run on hardware
//
#include <hip/hip_runtime.h>
#include <math.h>

#define BB 2
#define TT 512
#define BTOT (BB*TT)          // 1024 token rows
#define DD 768
#define FFD 1536
#define VV 50257
#define NWIN 4
#define KRET 8
#define NSTEPS 6
#define LDK 40                // padded LDS row stride (bf16 elems): 80B, conflict-free

typedef __attribute__((ext_vector_type(16))) __bf16 v16bf;
typedef __attribute__((ext_vector_type(8)))  __bf16 v8bf;
typedef __attribute__((ext_vector_type(4)))  __bf16 v4bf;
typedef __attribute__((ext_vector_type(8)))  float  v8f;

__device__ __forceinline__ float siluf(float x){ return x / (1.f + expf(-x)); }
__device__ __forceinline__ float softplusf(float x){ return (x > 20.f) ? x : log1pf(expf(x)); }
__device__ __forceinline__ float sigmf(float x){ return 1.f / (1.f + expf(-x)); }

// 4-element load + convert helpers (overload on storage type)
__device__ __forceinline__ v4bf ld4(const float* p){
  float4 f = *(const float4*)p;
  v4bf r; r[0] = (__bf16)f.x; r[1] = (__bf16)f.y; r[2] = (__bf16)f.z; r[3] = (__bf16)f.w;
  return r;
}
__device__ __forceinline__ v4bf ld4(const __bf16* p){ return *(const v4bf*)p; }

// ---------------------------------------------------------------------------
// Generic WMMA GEMM: C[M,N] = ACT(scale * (A @ W) + bias)
// A: fp32 [M,K] (lda, K%4==0, 16B-aligned rows for the WMMA path).
// W: fp32/bf16, [K,N] (ldw) or, if TRANSB, [N,K] (ldw).
// bf16 16x16x32 WMMA, fp32 accumulate. Block = 128 thr (4 waves), tile 64x64.
// LDS: As[m][k] and Bt[n][k], both k-contiguous so fragments load as b128 pairs.
// ---------------------------------------------------------------------------
template<int ACT, bool TRANSB, typename WT>
__global__ __launch_bounds__(128)
void wmma_gemm(const float* __restrict__ A, const WT* __restrict__ W,
               const float* __restrict__ bias, float* __restrict__ C,
               int M, int N, int K, int lda, int ldw, int ldc,
               long long sA, long long sW, long long sC, float scale)
{
  __shared__ __align__(16) __bf16 As[64 * LDK];
  __shared__ __align__(16) __bf16 Bt[64 * LDK];
  const int tid  = threadIdx.x;
  const int wave = tid >> 5;
  const int lane = tid & 31;
  const int mBase = blockIdx.y * 64;
  const int nBase = blockIdx.x * 64;
  A += (long long)blockIdx.z * sA;
  W += (long long)blockIdx.z * sW;
  C += (long long)blockIdx.z * sC;

  v8f acc[4];
  #pragma unroll
  for (int nt = 0; nt < 4; ++nt)
    #pragma unroll
    for (int r = 0; r < 8; ++r) acc[nt][r] = 0.f;

  const int mfr = wave * 16 + (lane & 15);     // A fragment row
  const int kh  = (lane >> 4) * 8;             // A fragment k-half offset
  const int kb2 = (lane >> 4) * 16;            // unused: B fragment uses kh too

  for (int k0 = 0; k0 < K; k0 += 32) {
    // ---- stage A tile (64 rows x 32 k), vectorized, fp32 -> bf16 ----
    #pragma unroll
    for (int g = tid; g < 512; g += 128) {
      int m  = g >> 3;
      int kq = (g & 7) * 4;
      int mm = mBase + m, kk = k0 + kq;
      v4bf bv;
      if (mm < M && kk + 3 < K) {
        bv = ld4(A + (long long)mm * lda + kk);
        if (kk + 32 < K)  // prefetch next k-chunk of A (global_prefetch_b8)
          __builtin_prefetch(A + (long long)mm * lda + kk + 32, 0, 1);
      } else {
        bv[0] = (__bf16)0.f; bv[1] = (__bf16)0.f; bv[2] = (__bf16)0.f; bv[3] = (__bf16)0.f;
      }
      *(v4bf*)(As + m * LDK + kq) = bv;
    }
    // ---- stage B tile transposed into Bt[n][k] ----
    if (TRANSB) {
      // W is [N,K]: straight k-contiguous loads
      #pragma unroll
      for (int g = tid; g < 512; g += 128) {
        int n  = g >> 3;
        int kq = (g & 7) * 4;
        int nn = nBase + n, kk = k0 + kq;
        v4bf bv;
        if (nn < N && kk + 3 < K) {
          bv = ld4(W + (long long)nn * ldw + kk);
        } else {
          bv[0] = (__bf16)0.f; bv[1] = (__bf16)0.f; bv[2] = (__bf16)0.f; bv[3] = (__bf16)0.f;
        }
        *(v4bf*)(Bt + n * LDK + kq) = bv;
      }
    } else {
      // W is [K,N]: gather 4 k-values at lane-consecutive n (coalesced dwords)
      #pragma unroll
      for (int g = tid; g < 512; g += 128) {
        int n  = g & 63;
        int kq = (g >> 6) * 4;
        int nn = nBase + n;
        v4bf bv;
        #pragma unroll
        for (int j = 0; j < 4; ++j) {
          int kk = k0 + kq + j;
          float v = (nn < N && kk < K) ? (float)W[(long long)kk * ldw + nn] : 0.f;
          bv[j] = (__bf16)v;
        }
        *(v4bf*)(Bt + n * LDK + kq) = bv;
      }
    }
    __syncthreads();

    // ---- A fragment: two contiguous b128 loads ----
    v8bf a0 = *(const v8bf*)(As + mfr * LDK + kh);
    v8bf a1 = *(const v8bf*)(As + mfr * LDK + 16 + kh);
    v16bf af = __builtin_shufflevector(a0, a1, 0,1,2,3,4,5,6,7,8,9,10,11,12,13,14,15);

    #pragma unroll
    for (int nt = 0; nt < 4; ++nt) {
      int n = nt * 16 + (lane & 15);
      v8bf b0 = *(const v8bf*)(Bt + n * LDK + kh);
      v8bf b1 = *(const v8bf*)(Bt + n * LDK + 16 + kh);
      v16bf bf = __builtin_shufflevector(b0, b1, 0,1,2,3,4,5,6,7,8,9,10,11,12,13,14,15);
      acc[nt] = __builtin_amdgcn_wmma_f32_16x16x32_bf16(
          false, af, false, bf, (short)0, acc[nt], false, false);
    }
    __syncthreads();
  }

  // store: C/D layout: lane -> n = lane&15, m = r + 8*(lane>>4)
  int mrow = mBase + wave * 16 + (lane >> 4) * 8;
  #pragma unroll
  for (int nt = 0; nt < 4; ++nt) {
    int n = nBase + nt * 16 + (lane & 15);
    if (n >= N) continue;
    float bv = bias ? bias[n] : 0.f;
    #pragma unroll
    for (int r = 0; r < 8; ++r) {
      int m = mrow + r;
      if (m >= M) continue;
      float v = acc[nt][r] * scale + bv;
      if      (ACT == 1) v = siluf(v);
      else if (ACT == 2) v = softplusf(v);
      else if (ACT == 3) v = sigmf(v);
      C[(long long)m * ldc + n] = v;
    }
  }
}

template<int ACT, bool TB, typename WT>
static void gemm(hipStream_t s, const float* A, const WT* W, const float* bias,
                 float* C, int M, int N, int K, int lda, int ldw, int ldc,
                 long long sA, long long sW, long long sC, int batch, float scale)
{
  dim3 g((N + 63) / 64, (M + 63) / 64, batch);
  wmma_gemm<ACT, TB, WT><<<g, 128, 0, s>>>(A, W, bias, C, M, N, K, lda, ldw, ldc,
                                           sA, sW, sC, scale);
}

// Tiny-N projections: one thread per output element (VALU path).
template<int ACT>
__global__ void small_gemm(const float* __restrict__ A, const float* __restrict__ W,
                           const float* __restrict__ bias, float* __restrict__ C,
                           int M, int N, int K)
{
  long long i = (long long)blockIdx.x * 256 + threadIdx.x;
  if (i >= (long long)M * N) return;
  int m = (int)(i / N), n = (int)(i % N);
  float s = 0.f;
  for (int k = 0; k < K; ++k) s += A[(long long)m * K + k] * W[(long long)k * N + n];
  if (bias) s += bias[n];
  if      (ACT == 1) s = siluf(s);
  else if (ACT == 3) s = sigmf(s);
  C[i] = s;
}

// ------------------------- elementwise / reductions -------------------------
__global__ void k_zero(float* p, long long n){
  long long i = (long long)blockIdx.x * 256 + threadIdx.x; if (i < n) p[i] = 0.f;
}
__global__ void k_addc(float* p, float c, long long n){
  long long i = (long long)blockIdx.x * 256 + threadIdx.x; if (i < n) p[i] += c;
}
__global__ void k_embed(float* h, const float* emb, const float* pos, const int* x){
  long long i = (long long)blockIdx.x * 256 + threadIdx.x;
  if (i >= (long long)BTOT * DD) return;
  int bt = (int)(i / DD), d = (int)(i % DD);
  int t = bt % TT;
  h[i] = emb[(long long)x[bt] * DD + d] + pos[(long long)t * DD + d];
}
__global__ void k_cvt_emb(__bf16* eB, const float* emb){
  long long i = (long long)blockIdx.x * 256 + threadIdx.x;
  if (i >= (long long)VV * DD) return;
  eB[i] = (__bf16)emb[i];
}
__global__ void k_init_state(float* pi, float* alive, float* pv, float* prr){
  long long i = (long long)blockIdx.x * 256 + threadIdx.x;
  if (i < (long long)BTOT * 7) pi[i] = ((i % 7) == 2) ? 1.f : 0.f;
  if (i < BTOT){ alive[i] = 1.f; pv[i] = 0.f; }
  if (i < (long long)BTOT * DD) prr[i] = 0.f;
}
__global__ __launch_bounds__(256) void k_row_stat(float* stat, const float* lam){
  __shared__ float sm[256];
  int bt = blockIdx.x, tid = threadIdx.x;
  float s = 0.f;
  for (int d = tid; d < DD; d += 256) s += lam[(long long)bt * DD + d];
  sm[tid] = s; __syncthreads();
  for (int o = 128; o > 0; o >>= 1){ if (tid < o) sm[tid] += sm[tid + o]; __syncthreads(); }
  if (tid == 0) stat[bt] = logf(fmaxf(sm[0] / DD, 1e-6f));
}
__global__ void k_build_gin(float* gin, const float* mu, const float* stat, const float* pv){
  long long i = (long long)blockIdx.x * 256 + threadIdx.x;
  if (i >= (long long)BTOT * (DD + 2)) return;
  int bt = (int)(i / (DD + 2)), c = (int)(i % (DD + 2));
  gin[i] = (c < DD) ? mu[(long long)bt * DD + c] : (c == DD ? stat[bt] : pv[bt]);
}
__global__ void k_trans(float* piev, const float* base49, const float* gate4,
                        const float* ML, const float* pi){
  int bt = blockIdx.x * blockDim.x + threadIdx.x; if (bt >= BTOT) return;
  const unsigned char SG[49] = {1,1,1,0,0,0,0, 0,1,1,1,0,0,0, 0,0,1,1,1,0,0,
                                0,0,0,1,1,1,1, 0,0,0,1,1,1,1, 0,0,0,1,0,1,1,
                                0,0,0,1,0,0,1};
  float g[4]; float mx = -1e30f;
  for (int m = 0; m < 4; ++m){ g[m] = gate4[bt * 4 + m]; mx = fmaxf(mx, g[m]); }
  float sum = 0.f;
  for (int m = 0; m < 4; ++m){ g[m] = expf(g[m] - mx); sum += g[m]; }
  for (int m = 0; m < 4; ++m) g[m] /= sum;
  float pe[7] = {0,0,0,0,0,0,0};
  for (int ii = 0; ii < 7; ++ii) {
    float row[7]; float rmx = -1e30f;
    for (int j = 0; j < 7; ++j) {
      if (!SG[ii*7+j]) { row[j] = -1e30f; continue; }
      float v = base49[(long long)bt*49 + ii*7 + j];
      for (int m = 0; m < 4; ++m) v += g[m] * ML[m*49 + ii*7 + j];
      row[j] = v; rmx = fmaxf(rmx, v);
    }
    float rs = 0.f;
    for (int j = 0; j < 7; ++j){
      if (SG[ii*7+j]) { row[j] = expf(row[j] - rmx); rs += row[j]; } else row[j] = 0.f;
    }
    float pii = pi[bt*7 + ii];
    for (int j = 0; j < 7; ++j) pe[j] += pii * (row[j] / rs);
  }
  for (int j = 0; j < 7; ++j) piev[bt*7 + j] = pe[j];
}
__global__ void k_scale_accum(float* dst, const float* src, const float* pi, int s){
  long long i = (long long)blockIdx.x * 256 + threadIdx.x;
  if (i >= (long long)BTOT * DD) return;
  int bt = (int)(i / DD);
  dst[i] += pi[bt*7 + s] * src[i];
}
__global__ void k_pi_update(float* pi, const float* piev, const float* ev,
                            const float* alive, float* rgate, float* ap4){
  int bt = blockIdx.x * blockDim.x + threadIdx.x; if (bt >= BTOT) return;
  float e[7]; float mx = -1e30f;
  for (int j = 0; j < 7; ++j){ e[j] = ev[bt*7+j] * 2.f; mx = fmaxf(mx, e[j]); }
  float s = 0.f;
  for (int j = 0; j < 7; ++j){ e[j] = expf(e[j] - mx); s += e[j]; }
  float pn[7]; float s2 = 0.f;
  for (int j = 0; j < 7; ++j){ pn[j] = piev[bt*7+j] * (e[j] / s); s2 += pn[j]; }
  s2 = fmaxf(s2, 1e-8f);
  for (int j = 0; j < 7; ++j) pn[j] /= s2;
  int i1 = 0;
  for (int j = 1; j < 7; ++j) if (pn[j] > pn[i1]) i1 = j;
  int i2 = -1;
  for (int j = 0; j < 7; ++j){ if (j == i1) continue; if (i2 < 0 || pn[j] > pn[i2]) i2 = j; }
  float tot = fmaxf(pn[i1] + pn[i2], 1e-8f);
  for (int j = 0; j < 7; ++j)
    pi[bt*7+j] = (j == i1 || j == i2) ? pn[j] / tot : 0.f;
  rgate[bt] = alive[bt] * pi[bt*7 + 3];
  ap4[bt]   = alive[bt] * pi[bt*7 + 4];
}
__global__ void k_build_cat(float* cat, const float* mu){
  long long i = (long long)blockIdx.x * 256 + threadIdx.x;
  long long n = (long long)BTOT * NWIN * (2*DD); if (i >= n) return;
  long long row = i / (2*DD); int c = (int)(i % (2*DD));
  int w = (int)(row % NWIN); long long bt = row / NWIN;
  int b = (int)(bt / TT), t = (int)(bt % TT);
  float v;
  if (c < DD) v = mu[bt*DD + c];
  else { int ts = t - (w + 1);
         v = (ts >= 0) ? mu[((long long)b*TT + ts)*DD + (c - DD)] : 0.f; }
  cat[i] = v;
}
__global__ void k_mean4(float* lmsg, const float* mo){
  long long i = (long long)blockIdx.x * 256 + threadIdx.x;
  if (i >= (long long)BTOT * DD) return;
  long long bt = i / DD; int d = (int)(i % DD);
  float s = 0.f;
  for (int w = 0; w < NWIN; ++w) s += mo[(bt*NWIN + w)*DD + d];
  lmsg[i] = s * 0.25f;
}
__global__ void k_concat2(float* dst, const float* a, const float* b, int d1, int d2, int M){
  long long i = (long long)blockIdx.x * 256 + threadIdx.x;
  long long n = (long long)M * (d1 + d2); if (i >= n) return;
  int row = (int)(i / (d1 + d2)), c = (int)(i % (d1 + d2));
  dst[i] = (c < d1) ? a[(long long)row*d1 + c] : b[(long long)row*d2 + (c - d1)];
}
__global__ void k_msg_gate(float* msgs, const float* rgate, const float* pv, const float* prr){
  long long i = (long long)blockIdx.x * 256 + threadIdx.x;
  if (i >= (long long)BTOT * DD) return;
  long long bt = i / DD;
  msgs[i] = rgate[bt] * (msgs[i] + 0.3f * (1.f - pv[bt]) * prr[i]);
}
__global__ void k_bayes(float* mu, float* lam, const float* m, const float* kap,
                        const float* ap4, const float* alive, const float* so){
  long long i = (long long)blockIdx.x * 256 + threadIdx.x;
  if (i >= (long long)BTOT * DD) return;
  long long bt = i / DD;
  float eff = fminf(ap4[bt] * kap[i], 10.f);
  float ln  = lam[i] + eff;
  float nmu = (lam[i] * mu[i] + eff * m[i]) / fmaxf(ln, 1e-6f);
  lam[i] = ln;
  mu[i]  = nmu + alive[bt] * so[i] * 0.1f;
}
__global__ __launch_bounds__(256) void k_layernorm(float* xn, const float* mu,
                                                   const float* g, const float* b){
  __shared__ float s1[256], s2[256];
  int bt = blockIdx.x, tid = threadIdx.x;
  float a = 0.f, q = 0.f;
  for (int d = tid; d < DD; d += 256){
    float v = mu[(long long)bt*DD + d]; a += v; q += v*v;
  }
  s1[tid] = a; s2[tid] = q; __syncthreads();
  for (int o = 128; o > 0; o >>= 1){
    if (tid < o){ s1[tid] += s1[tid+o]; s2[tid] += s2[tid+o]; } __syncthreads();
  }
  float mean = s1[0] / DD;
  float var  = s2[0] / DD - mean * mean;
  float inv  = rsqrtf(var + 1e-5f);
  for (int d = tid; d < DD; d += 256){
    float v = (mu[(long long)bt*DD + d] - mean) * inv;
    xn[(long long)bt*DD + d] = v * g[d] + b[d];
  }
}
__global__ __launch_bounds__(256) void k_argmax(int* idx, const float* logits){
  __shared__ float sv[256]; __shared__ int si[256];
  int bt = blockIdx.x, tid = threadIdx.x;
  float best = -1e30f; int bi = 0;
  for (int v = tid; v < VV; v += 256){
    float s = logits[(long long)bt*VV + v];
    if (s > best){ best = s; bi = v; }
  }
  sv[tid] = best; si[tid] = bi; __syncthreads();
  for (int o = 128; o > 0; o >>= 1){
    if (tid < o){
      if (sv[tid+o] > sv[tid] || (sv[tid+o] == sv[tid] && si[tid+o] < si[tid])){
        sv[tid] = sv[tid+o]; si[tid] = si[tid+o];
      }
    }
    __syncthreads();
  }
  if (tid == 0) idx[bt] = si[0];
}
__global__ void k_gather_emb(float* pe, const float* emb, const int* idx){
  long long i = (long long)blockIdx.x * 256 + threadIdx.x;
  if (i >= (long long)BTOT * DD) return;
  long long bt = i / DD; int d = (int)(i % DD);
  pe[i] = emb[(long long)idx[bt] * DD + d];
}
__global__ void k_sub(float* dst, const float* a, const float* b, long long n){
  long long i = (long long)blockIdx.x * 256 + threadIdx.x;
  if (i < n) dst[i] = a[i] - b[i];
}
__global__ void k_halt(float* halt, const float* pre, const float* alive, int isLast){
  int bt = blockIdx.x * blockDim.x + threadIdx.x; if (bt >= BTOT) return;
  halt[bt] = isLast ? alive[bt] : alive[bt] * sigmf(pre[bt]);
}
__global__ void k_accum(float* out, const float* logits, const float* halt){
  long long i = (long long)blockIdx.x * 256 + threadIdx.x;
  if (i >= (long long)BTOT * VV) return;
  out[i] += halt[i / VV] * logits[i];
}
__global__ void k_alive(float* alive, const float* halt){
  int bt = blockIdx.x * blockDim.x + threadIdx.x; if (bt >= BTOT) return;
  alive[bt] *= (1.f - halt[bt]);
}
// Causal top-8 sparse attention: one block (256 thr) per (b,t) row.
__global__ __launch_bounds__(256) void k_topk_attn(float* aout, const float* scores,
                                                   const float* vmat){
  __shared__ float rv[256]; __shared__ int ri[256];
  __shared__ int   cidx[KRET]; __shared__ float cw[KRET];
  int row = blockIdx.x, tid = threadIdx.x;
  int b = row / TT, t = row % TT;
  const float* srow = scores + ((long long)b*TT + t) * TT;
  for (int it = 0; it < KRET; ++it) {
    float best = -1e30f; int bi = -1;
    for (int j = tid; j < TT; j += 256) {
      if (j > t) continue;
      bool taken = false;
      for (int q = 0; q < it; ++q) if (cidx[q] == j) taken = true;
      if (taken) continue;
      float s = srow[j];
      if (s > best){ best = s; bi = j; }
    }
    rv[tid] = best; ri[tid] = bi; __syncthreads();
    for (int o = 128; o > 0; o >>= 1){
      if (tid < o && rv[tid+o] > rv[tid]){ rv[tid] = rv[tid+o]; ri[tid] = ri[tid+o]; }
      __syncthreads();
    }
    if (tid == 0){ cidx[it] = (ri[0] < 0) ? 0 : ri[0]; cw[it] = rv[0]; }
    __syncthreads();
  }
  if (tid == 0) {
    float mx = cw[0], s = 0.f, w[KRET];
    for (int q = 0; q < KRET; ++q){ w[q] = expf(cw[q] - mx); s += w[q]; }
    for (int q = 0; q < KRET; ++q) cw[q] = w[q] / s;
  }
  __syncthreads();
  for (int d = tid; d < DD; d += 256) {
    float s = 0.f;
    for (int q = 0; q < KRET; ++q)
      s += cw[q] * vmat[((long long)b*TT + cidx[q]) * DD + d];
    aout[(long long)row*DD + d] = s;
  }
}

static inline unsigned gblk(long long n){ return (unsigned)((n + 255) / 256); }

extern "C" void kernel_launch(void* const* d_in, const int* in_sizes, int n_in,
                              void* d_out, int out_size, void* d_ws, size_t ws_size,
                              hipStream_t stream)
{
  (void)in_sizes; (void)n_in; (void)out_size; (void)ws_size;
  const int*   x      = (const int*)  d_in[0];
  const float* emb    = (const float*)d_in[1];
  const float* pos    = (const float*)d_in[2];
  const float* imu_W  = (const float*)d_in[3];  const float* imu_b = (const float*)d_in[4];
  const float* ilam_W = (const float*)d_in[5];  const float* ilam_b= (const float*)d_in[6];
  const float* trb_W1 = (const float*)d_in[7];  const float* trb_b1= (const float*)d_in[8];
  const float* trb_W2 = (const float*)d_in[9];  const float* trb_b2= (const float*)d_in[10];
  const float* trg_W1 = (const float*)d_in[11]; const float* trg_b1= (const float*)d_in[12];
  const float* trg_W2 = (const float*)d_in[13]; const float* trg_b2= (const float*)d_in[14];
  const float* mlg    = (const float*)d_in[15];
  const float* sb_W1  = (const float*)d_in[16]; const float* sb_b1 = (const float*)d_in[17];
  const float* sb_W2  = (const float*)d_in[18]; const float* sb_b2 = (const float*)d_in[19];
  const float* ev_W   = (const float*)d_in[20]; const float* ev_b  = (const float*)d_in[21];
  const float* q_W    = (const float*)d_in[22]; const float* q_b   = (const float*)d_in[23];
  const float* kk_W   = (const float*)d_in[24]; const float* kk_b  = (const float*)d_in[25];
  const float* v_W    = (const float*)d_in[26]; const float* v_b   = (const float*)d_in[27];
  const float* msg_W1 = (const float*)d_in[28]; const float* msg_b1= (const float*)d_in[29];
  const float* msg_W2 = (const float*)d_in[30]; const float* msg_b2= (const float*)d_in[31];
  const float* out_W  = (const float*)d_in[32]; const float* out_b = (const float*)d_in[33];
  const float* wrm_W  = (const float*)d_in[34]; const float* wrm_b = (const float*)d_in[35];
  const float* wrg_W  = (const float*)d_in[36]; const float* wrg_b = (const float*)d_in[37];
  const float* vf_W1  = (const float*)d_in[38]; const float* vf_b1 = (const float*)d_in[39];
  const float* vf_W2  = (const float*)d_in[40]; const float* vf_b2 = (const float*)d_in[41];
  const float* rr_W   = (const float*)d_in[42]; const float* rr_b  = (const float*)d_in[43];
  const float* ha_W1  = (const float*)d_in[44]; const float* ha_b1 = (const float*)d_in[45];
  const float* ha_W2  = (const float*)d_in[46]; const float* ha_b2 = (const float*)d_in[47];
  const float* ln_g   = (const float*)d_in[48]; const float* ln_b  = (const float*)d_in[49];
  float* out = (float*)d_out;

  // ---- workspace bump allocator ----
  char* wp = (char*)d_ws;
  auto take = [&](size_t bytes) -> void* {
    void* r = (void*)wp; wp += (bytes + 255) & ~(size_t)255; return r;
  };
  __bf16* embB = (__bf16*)take((size_t)VV * DD * sizeof(__bf16));  // 77 MB (fits L2)
  float* logits = (float*)take((size_t)BTOT * VV * 4);
  float* h    = (float*)take((size_t)BTOT*DD*4);
  float* mu   = (float*)take((size_t)BTOT*DD*4);
  float* lam  = (float*)take((size_t)BTOT*DD*4);
  float* so   = (float*)take((size_t)BTOT*DD*4);
  float* tD0  = (float*)take((size_t)BTOT*DD*4);
  float* tD1  = (float*)take((size_t)BTOT*DD*4);
  float* qb   = (float*)take((size_t)BTOT*DD*4);
  float* kb   = (float*)take((size_t)BTOT*DD*4);
  float* vb   = (float*)take((size_t)BTOT*DD*4);
  float* lmsg = (float*)take((size_t)BTOT*DD*4);
  float* aout = (float*)take((size_t)BTOT*DD*4);
  float* msgs = (float*)take((size_t)BTOT*DD*4);
  float* mW   = (float*)take((size_t)BTOT*DD*4);
  float* kap  = (float*)take((size_t)BTOT*DD*4);
  float* pemb = (float*)take((size_t)BTOT*DD*4);
  float* xn   = (float*)take((size_t)BTOT*DD*4);
  float* prr  = (float*)take((size_t)BTOT*DD*4);
  float* t1536= (float*)take((size_t)BTOT*FFD*4);
  float* catW = (float*)take((size_t)BTOT*NWIN*2*DD*4);
  float* mhid = (float*)take((size_t)BTOT*NWIN*DD*4);
  float* mout = (float*)take((size_t)BTOT*NWIN*DD*4);
  float* gin  = (float*)take((size_t)BTOT*(DD+2)*4);
  float* sc   = (float*)take((size_t)BB*TT*TT*4);
  float* t256 = (float*)take((size_t)BTOT*256*4);
  float* base49=(float*)take((size_t)BTOT*49*4);
  float* g64  = (float*)take((size_t)BTOT*64*4);
  float* gate4= (float*)take((size_t)BTOT*4*4);
  float* ev7  = (float*)take((size_t)BTOT*7*4);
  float* pi   = (float*)take((size_t)BTOT*7*4);
  float* piev = (float*)take((size_t)BTOT*7*4);
  float* stat = (float*)take((size_t)BTOT*4);
  float* pv   = (float*)take((size_t)BTOT*4);
  float* alive= (float*)take((size_t)BTOT*4);
  float* halt = (float*)take((size_t)BTOT*4);
  float* rgate= (float*)take((size_t)BTOT*4);
  float* ap4  = (float*)take((size_t)BTOT*4);
  float* hpre = (float*)take((size_t)BTOT*4);
  int*   pidx = (int*)  take((size_t)BTOT*4);

  const float inv_sqrt_d = 1.f / sqrtf((float)DD);
  const long long nBTD = (long long)BTOT*DD;

  // ---- init ----
  k_zero<<<gblk((long long)BTOT*VV), 256, 0, stream>>>(out, (long long)BTOT*VV);
  k_embed<<<gblk(nBTD), 256, 0, stream>>>(h, emb, pos, x);
  k_cvt_emb<<<gblk((long long)VV*DD), 256, 0, stream>>>(embB, emb);
  k_init_state<<<gblk(nBTD), 256, 0, stream>>>(pi, alive, pv, prr);
  gemm<0,false,float>(stream, h, imu_W,  imu_b,  mu,  BTOT, DD, DD, DD, DD, DD, 0,0,0, 1, 1.f);
  gemm<2,false,float>(stream, h, ilam_W, ilam_b, lam, BTOT, DD, DD, DD, DD, DD, 0,0,0, 1, 1.f);
  k_addc<<<gblk(nBTD), 256, 0, stream>>>(lam, 0.1f, nBTD);

  // ---- recurrence ----
  for (int t = 0; t < NSTEPS; ++t) {
    // transition kernel
    k_row_stat<<<BTOT, 256, 0, stream>>>(stat, lam);
    k_build_gin<<<gblk((long long)BTOT*(DD+2)), 256, 0, stream>>>(gin, mu, stat, pv);
    gemm<1,false,float>(stream, mu, trb_W1, trb_b1, t256, BTOT, 256, DD, DD, 256, 256, 0,0,0, 1, 1.f);
    small_gemm<0><<<gblk((long long)BTOT*49), 256, 0, stream>>>(t256, trb_W2, trb_b2, base49, BTOT, 49, 256);
    small_gemm<1><<<gblk((long long)BTOT*64), 256, 0, stream>>>(gin, trg_W1, trg_b1, g64, BTOT, 64, DD+2);
    small_gemm<0><<<gblk((long long)BTOT*4),  256, 0, stream>>>(g64, trg_W2, trg_b2, gate4, BTOT, 4, 64);
    k_trans<<<gblk(BTOT), 256, 0, stream>>>(piev, base49, gate4, mlg, pi);

    // stage bank (7 experts, pi-weighted)
    k_zero<<<gblk(nBTD), 256, 0, stream>>>(so, nBTD);
    for (int s = 0; s < 7; ++s) {
      gemm<1,false,float>(stream, mu, sb_W1 + (long long)s*DD*FFD, sb_b1 + s*FFD,
                          t1536, BTOT, FFD, DD, DD, FFD, FFD, 0,0,0, 1, 1.f);
      gemm<0,false,float>(stream, t1536, sb_W2 + (long long)s*FFD*DD, sb_b2 + s*DD,
                          tD0, BTOT, DD, FFD, FFD, DD, DD, 0,0,0, 1, 1.f);
      k_scale_accum<<<gblk(nBTD), 256, 0, stream>>>(so, tD0, pi, s);
    }
    small_gemm<0><<<gblk((long long)BTOT*7), 256, 0, stream>>>(so, ev_W, ev_b, ev7, BTOT, 7, DD);
    k_pi_update<<<gblk(BTOT), 256, 0, stream>>>(pi, piev, ev7, alive, rgate, ap4);

    // local router: windowed messages
    k_build_cat<<<gblk((long long)BTOT*NWIN*2*DD), 256, 0, stream>>>(catW, mu);
    gemm<1,false,float>(stream, catW, msg_W1, msg_b1, mhid, BTOT*NWIN, DD, 2*DD, 2*DD, DD, DD, 0,0,0, 1, 1.f);
    gemm<0,false,float>(stream, mhid, msg_W2, msg_b2, mout, BTOT*NWIN, DD, DD, DD, DD, DD, 0,0,0, 1, 1.f);
    k_mean4<<<gblk(nBTD), 256, 0, stream>>>(lmsg, mout);

    // causal top-8 sparse attention
    gemm<0,false,float>(stream, mu, q_W,  q_b,  qb, BTOT, DD, DD, DD, DD, DD, 0,0,0, 1, 1.f);
    gemm<0,false,float>(stream, mu, kk_W, kk_b, kb, BTOT, DD, DD, DD, DD, DD, 0,0,0, 1, 1.f);
    gemm<0,false,float>(stream, mu, v_W,  v_b,  vb, BTOT, DD, DD, DD, DD, DD, 0,0,0, 1, 1.f);
    gemm<0,true,float>(stream, qb, kb, nullptr, sc, TT, TT, DD, DD, DD, TT,
                       (long long)TT*DD, (long long)TT*DD, (long long)TT*TT, BB, inv_sqrt_d);
    k_topk_attn<<<BTOT, 256, 0, stream>>>(aout, sc, vb);

    // combine messages
    k_concat2<<<gblk((long long)BTOT*2*DD), 256, 0, stream>>>(t1536, lmsg, aout, DD, DD, BTOT);
    gemm<0,false,float>(stream, t1536, out_W, out_b, msgs, BTOT, DD, 2*DD, 2*DD, DD, DD, 0,0,0, 1, 1.f);
    k_msg_gate<<<gblk(nBTD), 256, 0, stream>>>(msgs, rgate, pv, prr);

    // bayesian write
    k_concat2<<<gblk((long long)BTOT*2*DD), 256, 0, stream>>>(t1536, mu, msgs, DD, DD, BTOT);
    gemm<0,false,float>(stream, t1536, wrm_W, wrm_b, mW,  BTOT, DD, 2*DD, 2*DD, DD, DD, 0,0,0, 1, 1.f);
    gemm<2,false,float>(stream, t1536, wrg_W, wrg_b, kap, BTOT, DD, 2*DD, 2*DD, DD, DD, 0,0,0, 1, 1.f);
    k_bayes<<<gblk(nBTD), 256, 0, stream>>>(mu, lam, mW, kap, ap4, alive, so);

    // readout (bf16 emb streams from L2; TRANSB path = contiguous v4bf loads)
    k_layernorm<<<BTOT, 256, 0, stream>>>(xn, mu, ln_g, ln_b);
    gemm<0,true,__bf16>(stream, xn, embB, nullptr, logits, BTOT, VV, DD, DD, DD, VV, 0,0,0, 1, inv_sqrt_d);
    k_argmax<<<BTOT, 256, 0, stream>>>(pidx, logits);
    k_gather_emb<<<gblk(nBTD), 256, 0, stream>>>(pemb, emb, pidx);

    // verify + reroute
    k_concat2<<<gblk((long long)BTOT*2*DD), 256, 0, stream>>>(t1536, mu, pemb, DD, DD, BTOT);
    gemm<1,false,float>(stream, t1536, vf_W1, vf_b1, tD0, BTOT, DD, 2*DD, 2*DD, DD, DD, 0,0,0, 1, 1.f);
    small_gemm<3><<<gblk(BTOT), 256, 0, stream>>>(tD0, vf_W2, vf_b2, pv, BTOT, 1, DD);
    k_sub<<<gblk(nBTD), 256, 0, stream>>>(tD1, mu, pemb, nBTD);
    gemm<0,false,float>(stream, tD1, rr_W, rr_b, prr, BTOT, DD, DD, DD, DD, DD, 0,0,0, 1, 1.f);

    // halting
    if (t < NSTEPS - 1) {
      k_row_stat<<<BTOT, 256, 0, stream>>>(stat, lam);
      k_build_gin<<<gblk((long long)BTOT*(DD+2)), 256, 0, stream>>>(gin, mu, stat, pv);
      small_gemm<1><<<gblk((long long)BTOT*64), 256, 0, stream>>>(gin, ha_W1, ha_b1, g64, BTOT, 64, DD+2);
      small_gemm<0><<<gblk(BTOT), 256, 0, stream>>>(g64, ha_W2, ha_b2, hpre, BTOT, 1, 64);
    }
    k_halt<<<gblk(BTOT), 256, 0, stream>>>(halt, hpre, alive, (t == NSTEPS - 1) ? 1 : 0);
    k_accum<<<gblk((long long)BTOT*VV), 256, 0, stream>>>(out, logits, halt);
    k_alive<<<gblk(BTOT), 256, 0, stream>>>(alive, halt);
  }
}
